// FusionAttention_36893769073076
// MI455X (gfx1250) — matrix operand order repaired and checked by reading, not compiled, not run on hardware
//
#include <hip/hip_runtime.h>
#include <hip/hip_bf16.h>
#include <stdint.h>
#include <stddef.h>

typedef __bf16 bf16;
typedef __attribute__((ext_vector_type(16))) __bf16 v16bf;
typedef __attribute__((ext_vector_type(8)))  float  v8f;

#define NBRANCH 3
#define BB      16
#define NTOK    197
#define NHEAD   12
#define DH      64
#define DIM     768
#define M_ROWS  (BB*NTOK)   /* 3152 = 197 exact 16-row tiles */
#define M_BLK2  99          /* ceil(3152/32): 32-row blocks per wave */
#define N_QKV   (3*DIM)     /* 2304 */
#define KEYS    589         /* 197 + 2*196 */
#define KPAD    592         /* 37*16 */
#define VPAD    608         /* 19*32 */
#define SCALE_F 0.125f      /* 64^-0.5 */

// ---------------------------------------------------------------------------
// WMMA fragment helpers (layouts per CDNA5 ISA 7.12.2, wave32)
// A 16x32 bf16: lanes 0-15 -> row=lane, K chunks at 0 and 16;
//               lanes 16-31 -> row=lane-16, K chunks at 8 and 24.
// B 32x16 bf16: lane&15 = column; lanes<16 hold K rows 0-7,16-23, lanes>=16 K 8-15,24-31
//               (two 8-K chunks per lane, as A; B stored pre-transposed).
// C 16x16 f32 : c[i] -> row = i + 8*(lane>>4), col = lane&15.
// ---------------------------------------------------------------------------
struct alignas(32) Frag32B { uint4 lo, hi; };

static __device__ inline v16bf ld_frag(const bf16* p0, const bf16* p1) {
  Frag32B t;
  t.lo = *reinterpret_cast<const uint4*>(p0);
  t.hi = *reinterpret_cast<const uint4*>(p1);
  return __builtin_bit_cast(v16bf, t);
}

static __device__ inline v8f zero8() {
  v8f z = {0.f,0.f,0.f,0.f,0.f,0.f,0.f,0.f};
  return z;
}

static __device__ inline v8f wmma_bf16(v16bf a, v16bf b, v8f c) {
  return __builtin_amdgcn_wmma_f32_16x16x32_bf16(false, a, false, b, (short)0, c, false, false);
}

// ---------------------------------------------------------------------------
// Precision staging kernels
// ---------------------------------------------------------------------------
__global__ void k_cast(const float* __restrict__ src, bf16* __restrict__ dst, int n) {
  for (int i = blockIdx.x*blockDim.x + threadIdx.x; i < n; i += gridDim.x*blockDim.x)
    dst[i] = (bf16)src[i];
}

// w[K][N] row-major (fp32) -> wT[N][K] (bf16), so WMMA B-frags read contiguous K.
__global__ void k_transpose(const float* __restrict__ w, bf16* __restrict__ wT, int K, int N) {
  int total = K*N;
  for (int i = blockIdx.x*blockDim.x + threadIdx.x; i < total; i += gridDim.x*blockDim.x) {
    int n = i / K, k = i - n*K;
    wT[i] = (bf16)w[(size_t)k*N + n];
  }
}

// Zero the key-padding regions of Kc (589..591) and Vt (589..607).
__global__ void k_padzero(bf16* __restrict__ kc, bf16* __restrict__ vt) {
  const int G = NBRANCH*BB*NHEAD; // 576 (branch,batch,head) groups
  int tid = blockIdx.x*blockDim.x + threadIdx.x;
  int stride = gridDim.x*blockDim.x;
  const int kcTot = G * (KPAD-KEYS) * DH;
  for (int i = tid; i < kcTot; i += stride) {
    int g = i / ((KPAD-KEYS)*DH); int r = i - g*((KPAD-KEYS)*DH);
    int j = KEYS + r / DH;        int d = r % DH;
    kc[(size_t)g*KPAD*DH + (size_t)j*DH + d] = (bf16)0.f;
  }
  const int vtTot = G * DH * (VPAD-KEYS);
  for (int i = tid; i < vtTot; i += stride) {
    int g = i / (DH*(VPAD-KEYS)); int r = i - g*(DH*(VPAD-KEYS));
    int d = r / (VPAD-KEYS);      int j = KEYS + r % (VPAD-KEYS);
    vt[(size_t)g*DH*VPAD + (size_t)d*VPAD + j] = (bf16)0.f;
  }
}

// ---------------------------------------------------------------------------
// QKV epilogue scatter for one 16x16 C tile column group (helper).
// Q -> qb[g][tok][d]; K -> concat Kc[g][j][d] (own + other branches' summary
// slots, CLS dropped); V -> transposed Vt[g][d][j] (same concat mapping).
// ---------------------------------------------------------------------------
static __device__ inline void qkv_scatter(
    float cv, int n, int m, int br, float bv,
    bf16* __restrict__ qb, bf16* __restrict__ kc, bf16* __restrict__ vt)
{
  const int region = n / DIM;            // 0=q 1=k 2=v
  const int h = (n % DIM) / DH;
  const int d = n % DH;
  const int b   = m / NTOK;
  const int tok = m - b*NTOK;
  const bf16 val = (bf16)(cv + bv);
  const int g_own = (br*BB + b)*NHEAD + h;
  if (region == 0) {
    qb[((size_t)g_own*NTOK + tok)*DH + d] = val;
  } else if (region == 1) {
    kc[(size_t)g_own*KPAD*DH + (size_t)tok*DH + d] = val;
    if (tok > 0) {
      #pragma unroll
      for (int o = 0; o < NBRANCH; ++o) {
        if (o == br) continue;
        const int pos = br - (br > o ? 1 : 0);
        const int j2  = NTOK + pos*(NTOK-1) + tok - 1;
        const int g_o = (o*BB + b)*NHEAD + h;
        kc[(size_t)g_o*KPAD*DH + (size_t)j2*DH + d] = val;
      }
    }
  } else {
    vt[(size_t)g_own*DH*VPAD + (size_t)d*VPAD + tok] = val;
    if (tok > 0) {
      #pragma unroll
      for (int o = 0; o < NBRANCH; ++o) {
        if (o == br) continue;
        const int pos = br - (br > o ? 1 : 0);
        const int j2  = NTOK + pos*(NTOK-1) + tok - 1;
        const int g_o = (o*BB + b)*NHEAD + h;
        vt[(size_t)g_o*DH*VPAD + (size_t)d*VPAD + j2] = val;
      }
    }
  }
}

// ---------------------------------------------------------------------------
// Fused QKV GEMM: one wave computes a 32x128 tile of x@Wqkv + b
// (two 16-row A tiles share every B fragment: 16 WMMAs per 20 b128 loads).
// ---------------------------------------------------------------------------
__global__ __launch_bounds__(32) void k_qkv_gemm(
    const bf16* __restrict__ xb, const bf16* __restrict__ wT,
    const float* __restrict__ b0, const float* __restrict__ b1, const float* __restrict__ b2,
    bf16* __restrict__ qb, bf16* __restrict__ kc, bf16* __restrict__ vt)
{
  constexpr int NT = 8;
  const int lane = threadIdx.x & 31;
  const int ln = lane & 15, hi = lane >> 4;
  const int br = blockIdx.z;
  const int mb = blockIdx.y;             // 0..98, 32-row blocks
  const int n0 = blockIdx.x * (NT*16);   // 128-col groups (align with q/k/v regions)
  const float* bias = (br==0) ? b0 : (br==1 ? b1 : b2);
  const bf16* xbb = xb + (size_t)br*M_ROWS*DIM;
  const bf16* wTb = wT + (size_t)br*N_QKV*DIM;

  const int r0 = mb*32;
  const bool has2 = (r0 + 16) < M_ROWS;  // last block has a single 16-row tile
  const bf16* arow0 = xbb + (size_t)(r0 + ln)*DIM;
  const bf16* arow1 = has2 ? (arow0 + (size_t)16*DIM) : arow0;

  v8f acc0[NT], acc1[NT];
  #pragma unroll
  for (int nt=0; nt<NT; ++nt) { acc0[nt] = zero8(); acc1[nt] = zero8(); }

  for (int k0 = 0; k0 < DIM; k0 += 32) {
    __builtin_prefetch(arow0 + k0 + 256, 0, 3);   // global_prefetch_b8 (near)
    v16bf a0 = ld_frag(arow0 + k0 + hi*8, arow0 + k0 + 16 + hi*8);
    v16bf a1 = ld_frag(arow1 + k0 + hi*8, arow1 + k0 + 16 + hi*8);
    #pragma unroll
    for (int nt=0; nt<NT; ++nt) {
      const bf16* bp = wTb + (size_t)(n0 + nt*16 + ln)*DIM + k0 + hi*8;
      v16bf bfr = ld_frag(bp, bp + 16);
      acc0[nt] = wmma_bf16(a0, bfr, acc0[nt]);
      acc1[nt] = wmma_bf16(a1, bfr, acc1[nt]);
    }
  }

  #pragma unroll
  for (int nt=0; nt<NT; ++nt) {
    const int n = n0 + nt*16 + ln;
    const float bv = bias[n];
    #pragma unroll
    for (int i=0; i<8; ++i) {
      const int mrow = i + 8*hi;
      qkv_scatter(acc0[nt][i], n, r0 + mrow, br, bv, qb, kc, vt);
      if (has2)
        qkv_scatter(acc1[nt][i], n, r0 + 16 + mrow, br, bv, qb, kc, vt);
    }
  }
}

// ---------------------------------------------------------------------------
// Attention: one wave per (branch, b, h, 16-query tile).
// S = (Q Kc^T)*scale -> LDS f32; row softmax; P -> bf16 LDS; O = P Vt^T via WMMA.
// ---------------------------------------------------------------------------
__global__ __launch_bounds__(32) void k_attn(
    const bf16* __restrict__ qb, const bf16* __restrict__ kc,
    const bf16* __restrict__ vt, bf16* __restrict__ ao)
{
  __shared__ float Ssc[16][VPAD];   // 38.0 KB
  __shared__ bf16  Pbf[16][VPAD];   // 19.0 KB

  const int lane = threadIdx.x & 31;
  const int ln = lane & 15, hi = lane >> 4;
  const int qt = blockIdx.x;                 // 0..12
  const int b  = blockIdx.y / NHEAD;
  const int h  = blockIdx.y % NHEAD;
  const int br = blockIdx.z;
  const int g  = (br*BB + b)*NHEAD + h;

  // hoisted Q fragments: 16 rows x 64 d, as two 16x32 A-frags (clamp tail rows)
  int row = qt*16 + ln; if (row > NTOK-1) row = NTOK-1;
  const bf16* qrow = qb + ((size_t)g*NTOK + row)*DH;
  v16bf aq0 = ld_frag(qrow + 0  + hi*8, qrow + 16 + hi*8);
  v16bf aq1 = ld_frag(qrow + 32 + hi*8, qrow + 48 + hi*8);

  // Phase 1: scores for all 592 (padded) keys, 16-key tiles
  const bf16* kcb = kc + (size_t)g*KPAD*DH;
  for (int kt = 0; kt < KPAD/16; ++kt) {
    const bf16* krow = kcb + (size_t)(kt*16 + ln)*DH;   // lane = key column
    v16bf bk0 = ld_frag(krow + 0  + hi*8, krow + 16 + hi*8);
    v16bf bk1 = ld_frag(krow + 32 + hi*8, krow + 48 + hi*8);
    v8f s = zero8();
    s = wmma_bf16(aq0, bk0, s);
    s = wmma_bf16(aq1, bk1, s);
    const int col = kt*16 + ln;
    #pragma unroll
    for (int i=0; i<8; ++i) {
      float v = s[i] * SCALE_F;
      if (col >= KEYS) v = -1e30f;
      Ssc[i + 8*hi][col] = v;
    }
  }
  __syncthreads();

  // Phase 2: row softmax (lane l owns row l), emit bf16 P with zeroed pad cols
  if (lane < 16) {
    float mx = -1e30f;
    for (int c = 0; c < KPAD; ++c) mx = fmaxf(mx, Ssc[lane][c]);
    float sum = 0.f;
    for (int c = 0; c < KPAD; ++c) { float e = __expf(Ssc[lane][c] - mx); Ssc[lane][c] = e; sum += e; }
    const float inv = 1.f / sum;
    for (int c = 0; c < KPAD; ++c) Pbf[lane][c] = (bf16)(Ssc[lane][c] * inv);
    for (int c = KPAD; c < VPAD; ++c) Pbf[lane][c] = (bf16)0.f;
  }
  __syncthreads();

  // Phase 3: O(16x64) = P(16x608) * V(608x64); A-frags from LDS, B-frags from Vt
  const bf16* vtb = vt + (size_t)g*DH*VPAD;
  v8f o[4];
  #pragma unroll
  for (int nt=0; nt<4; ++nt) o[nt] = zero8();
  const bf16* prow = &Pbf[ln][0];
  for (int k0 = 0; k0 < VPAD; k0 += 32) {
    v16bf ap = ld_frag(prow + k0 + hi*8, prow + k0 + 16 + hi*8);  // ds_load_b128
    #pragma unroll
    for (int nt=0; nt<4; ++nt) {
      const bf16* vp = vtb + (size_t)(nt*16 + ln)*VPAD + k0 + hi*8;
      v16bf bv = ld_frag(vp, vp + 16);
      o[nt] = wmma_bf16(ap, bv, o[nt]);
    }
  }

  // Store attention output (bf16) in [br][b][tok][h*64+d] for the proj GEMM
  #pragma unroll
  for (int nt=0; nt<4; ++nt) {
    const int d = nt*16 + ln;
    #pragma unroll
    for (int i=0; i<8; ++i) {
      const int tok = qt*16 + i + 8*hi;
      if (tok < NTOK)
        ao[((size_t)(br*BB + b)*NTOK + tok)*DIM + h*DH + d] = (bf16)o[nt][i];
    }
  }
}

// ---------------------------------------------------------------------------
// Projection GEMM: out(f32) = att_out(bf16) @ proj_wT(bf16) + bias.
// One wave computes a 32x64 tile (two A tiles share each B fragment).
// ---------------------------------------------------------------------------
__global__ __launch_bounds__(32) void k_proj_gemm(
    const bf16* __restrict__ ao, const bf16* __restrict__ wT,
    const float* __restrict__ b0, const float* __restrict__ b1, const float* __restrict__ b2,
    float* __restrict__ out)
{
  constexpr int NT = 4;
  const int lane = threadIdx.x & 31;
  const int ln = lane & 15, hi = lane >> 4;
  const int br = blockIdx.z;
  const int mb = blockIdx.y;             // 0..98, 32-row blocks
  const int n0 = blockIdx.x * (NT*16);
  const float* bias = (br==0) ? b0 : (br==1 ? b1 : b2);
  const bf16* ab  = ao + (size_t)br*M_ROWS*DIM;
  const bf16* wTb = wT + (size_t)br*DIM*DIM;
  float* ob = out + (size_t)br*M_ROWS*DIM;

  const int r0 = mb*32;
  const bool has2 = (r0 + 16) < M_ROWS;
  const bf16* arow0 = ab + (size_t)(r0 + ln)*DIM;
  const bf16* arow1 = has2 ? (arow0 + (size_t)16*DIM) : arow0;

  v8f acc0[NT], acc1[NT];
  #pragma unroll
  for (int nt=0; nt<NT; ++nt) { acc0[nt] = zero8(); acc1[nt] = zero8(); }

  for (int k0 = 0; k0 < DIM; k0 += 32) {
    __builtin_prefetch(arow0 + k0 + 256, 0, 3);
    v16bf a0 = ld_frag(arow0 + k0 + hi*8, arow0 + k0 + 16 + hi*8);
    v16bf a1 = ld_frag(arow1 + k0 + hi*8, arow1 + k0 + 16 + hi*8);
    #pragma unroll
    for (int nt=0; nt<NT; ++nt) {
      const bf16* bp = wTb + (size_t)(n0 + nt*16 + ln)*DIM + k0 + hi*8;
      v16bf bfr = ld_frag(bp, bp + 16);
      acc0[nt] = wmma_bf16(a0, bfr, acc0[nt]);
      acc1[nt] = wmma_bf16(a1, bfr, acc1[nt]);
    }
  }

  #pragma unroll
  for (int nt=0; nt<NT; ++nt) {
    const int n = n0 + nt*16 + ln;
    const float bv = bias[n];
    #pragma unroll
    for (int i=0; i<8; ++i) {
      const int mrow = i + 8*hi;
      ob[(size_t)(r0 + mrow)*DIM + n] = acc0[nt][i] + bv;
      if (has2)
        ob[(size_t)(r0 + 16 + mrow)*DIM + n] = acc1[nt][i] + bv;
    }
  }
}

// ---------------------------------------------------------------------------
// Host-side launch
// ---------------------------------------------------------------------------
extern "C" void kernel_launch(void* const* d_in, const int* in_sizes, int n_in,
                              void* d_out, int out_size, void* d_ws, size_t ws_size,
                              hipStream_t stream) {
  (void)in_sizes; (void)n_in; (void)out_size; (void)ws_size;
  // setup_inputs dict order: per branch i: x, qkv_w, qkv_b, proj_w, proj_b
  const float *x[3], *qw[3], *qbias[3], *pw[3], *pbias[3];
  for (int i = 0; i < 3; ++i) {
    x[i]     = (const float*)d_in[5*i + 0];
    qw[i]    = (const float*)d_in[5*i + 1];
    qbias[i] = (const float*)d_in[5*i + 2];
    pw[i]    = (const float*)d_in[5*i + 3];
    pbias[i] = (const float*)d_in[5*i + 4];
  }

  char* ws = (char*)d_ws;
  size_t off = 0;
  auto alloc = [&](size_t bytes) -> void* {
    void* p = ws + off;
    off += (bytes + 255) & ~(size_t)255;
    return p;
  };
  bf16* xb  = (bf16*)alloc((size_t)NBRANCH*M_ROWS*DIM*2);          // bf16 x
  bf16* wqT = (bf16*)alloc((size_t)NBRANCH*N_QKV*DIM*2);           // qkv_w^T
  bf16* wpT = (bf16*)alloc((size_t)NBRANCH*DIM*DIM*2);             // proj_w^T
  bf16* qb  = (bf16*)alloc((size_t)NBRANCH*BB*NHEAD*NTOK*DH*2);    // Q
  bf16* kc  = (bf16*)alloc((size_t)NBRANCH*BB*NHEAD*KPAD*DH*2);    // concat K
  bf16* vt  = (bf16*)alloc((size_t)NBRANCH*BB*NHEAD*DH*VPAD*2);    // concat V^T
  bf16* ao  = (bf16*)alloc((size_t)NBRANCH*M_ROWS*DIM*2);          // attn out

  for (int i = 0; i < 3; ++i) {
    k_cast     <<<512,  256, 0, stream>>>(x[i],  xb  + (size_t)i*M_ROWS*DIM, M_ROWS*DIM);
    k_transpose<<<1024, 256, 0, stream>>>(qw[i], wqT + (size_t)i*N_QKV*DIM, DIM, N_QKV);
    k_transpose<<<512,  256, 0, stream>>>(pw[i], wpT + (size_t)i*DIM*DIM,   DIM, DIM);
  }
  k_padzero<<<512, 256, 0, stream>>>(kc, vt);

  k_qkv_gemm<<<dim3(N_QKV/128, M_BLK2, NBRANCH), 32, 0, stream>>>(
      xb, wqT, qbias[0], qbias[1], qbias[2], qb, kc, vt);

  k_attn<<<dim3((NTOK + 15)/16, BB*NHEAD, NBRANCH), 32, 0, stream>>>(qb, kc, vt, ao);

  k_proj_gemm<<<dim3(DIM/64, M_BLK2, NBRANCH), 32, 0, stream>>>(
      ao, wpT, pbias[0], pbias[1], pbias[2], (float*)d_out);
}
